// SparseAttentionBlock_56599079026971
// MI455X (gfx1250) — compile-verified, matrix operand order checked by reading, not run on hardware
//
#include <hip/hip_runtime.h>
#include <hip/hip_bf16.h>

// ---------------------------------------------------------------------------
// SparseAttentionBlock (linear attention) for MI455X / gfx1250, wave32 + WMMA.
// T=4 B=8 N=1024 C=512 H=8 D=64.  All GEMMs via v_wmma_f32_16x16x32_f16.
// BN folded into weights; intermediates kept in f16 to halve HBM traffic
// (bandwidth-bound: ~73 GFLOP vs ~0.5 GB traffic @ 23.3 TB/s).
// f16 LDS staging via GLOBAL_LOAD_ASYNC_TO_LDS_B128 (ASYNCcnt); the y-kernel's
// kv tile staged via the Tensor Data Mover (TENSORcnt).  All feature paths are
// __has_builtin-guarded with the proven round-1 fallback.
// ---------------------------------------------------------------------------

typedef __attribute__((ext_vector_type(8)))  _Float16 v8h;
typedef __attribute__((ext_vector_type(16))) _Float16 v16h;
typedef __attribute__((ext_vector_type(8)))  float    v8f;
typedef __attribute__((ext_vector_type(4)))  unsigned u32x4;
typedef __attribute__((ext_vector_type(4)))  int      i32x4;
typedef __attribute__((ext_vector_type(8)))  int      i32x8;

#define Cc   512
#define Nseq 1024
#define Mrows 32768          // T*B*N
#define CxC  262144          // 512*512
#define EPSf 1e-5f

#if __has_builtin(__builtin_amdgcn_global_load_async_to_lds_b128)
#define HAS_ASYNC_LDS 1
#endif
#if __has_builtin(__builtin_amdgcn_tensor_load_to_lds)
#define HAS_TDM 1
#endif

// Per-lane 16-byte global->LDS copy: async (ASYNCcnt) when available.
// Builtin signature is (int4 AS(1)* src, int4 AS(3)* dst, imm off, imm cpol).
__device__ inline void stage_b128(void* lds, const void* g) {
#ifdef HAS_ASYNC_LDS
    typedef __attribute__((address_space(1))) i32x4 gvec_t;
    typedef __attribute__((address_space(3))) i32x4 lvec_t;
    __builtin_amdgcn_global_load_async_to_lds_b128(
        (gvec_t*)(void*)g, (lvec_t*)lds, 0, 0);
#else
    *(v8h*)lds = *(const v8h*)g;
#endif
}

// Wait for this thread's async copies, then publish to the workgroup.
__device__ inline void stage_fence() {
#ifdef HAS_ASYNC_LDS
#if __has_builtin(__builtin_amdgcn_s_wait_asynccnt)
    __builtin_amdgcn_s_wait_asynccnt(0);
#else
    asm volatile("s_wait_asynccnt 0x0" ::: "memory");
#endif
#endif
    __syncthreads();
}

__device__ inline void wait_tensor0() {
#if __has_builtin(__builtin_amdgcn_s_wait_tensorcnt)
    __builtin_amdgcn_s_wait_tensorcnt(0);
#else
    asm volatile("s_wait_tensorcnt 0x0" ::: "memory");
#endif
}

struct AllParams {
    const float* W[4];
    const float* b[4];
    const float* g[4];
    const float* beta[4];
    const float* mu[4];
    const float* var[4];
};

// Workspace byte offsets (total ~132 MiB)
#define OFF_WPACK 0u
#define OFF_BIAS  2097152u
#define OFF_QH    2105344u
#define OFF_KT    35659776u
#define OFF_VH    69214208u
#define OFF_YH    102768640u
#define OFF_KVH   136323072u

// ---------------------------------------------------------------------------
// Kernel 1: fold BN into weights, pack f16 weights into per-lane WMMA
// B-fragment layout: tile (nt,kt) of 16(N)x32(K); lane<16 -> col=lane,
// K {0..7,16..23}; lane>=16 -> col=lane-16, K {8..15,24..31}. 16 halves/lane.
// ---------------------------------------------------------------------------
__global__ __launch_bounds__(256)
void fold_pack_kernel(AllParams P, _Float16* __restrict__ wpack,
                      float* __restrict__ biasF) {
    int gid = blockIdx.x * 256 + threadIdx.x;      // 0 .. 4*C*C-1
    int m = gid >> 18;
    int r = gid & (CxC - 1);
    int n = r >> 9;
    int k = r & 511;
    float s   = P.g[m][n] * rsqrtf(P.var[m][n] + EPSf);
    float val = P.W[m][n * Cc + k] * s;
    int nt = n >> 4, l0 = n & 15;
    int kt = k >> 5, kk = k & 31;
    int grp = kk >> 3, pos = kk & 7;
    int lane = l0 + ((grp & 1) << 4);
    int half = pos + ((grp >> 1) << 3);
    int idx  = ((nt * 16 + kt) * 32 + lane) * 16 + half;
    wpack[m * CxC + idx] = (_Float16)val;
    if (k == 0)
        biasF[m * Cc + n] = (P.b[m][n] - P.mu[m][n]) * s + P.beta[m][n];
}

__device__ inline v16h join16(v8h lo, v8h hi) {
    v16h a;
#pragma unroll
    for (int i = 0; i < 8; ++i) { a[i] = lo[i]; a[8 + i] = hi[i]; }
    return a;
}

// ---------------------------------------------------------------------------
// Kernel 2: q/k/v = x @ W'^T + b' (BN folded). 128x64 tile per block,
// 8 waves, each wave 16 rows x 64 cols x 3 matrices = 12 accums.
// k stored transposed per head for the kv GEMM.  fp32->f16 conversion keeps
// this staging on the VALU path (async copy cannot convert).
// ---------------------------------------------------------------------------
__global__ __launch_bounds__(256)
void qkv_kernel(const float* __restrict__ x, const _Float16* __restrict__ wpack,
                const float* __restrict__ biasF,
                _Float16* __restrict__ qh, _Float16* __restrict__ kT,
                _Float16* __restrict__ vh) {
    __shared__ __align__(16) _Float16 Xs[128 * 32];
    const int t = threadIdx.x, w = t >> 5, lane = t & 31;
    const int m0 = blockIdx.x * 128;
    const int nb = blockIdx.y;                     // 64-col tile
    const int laneRow = lane & 15;
    const int base0   = (lane < 16) ? 0 : 8;
    v8f acc[3][4] = {};

    for (int kc = 0; kc < 16; ++kc) {
        {   // stage 128x32 f32 -> f16 into LDS, coalesced float4 loads
            int tr = t >> 1, th = t & 1;
            const float4* xp =
                (const float4*)(x + (m0 + tr) * Cc + kc * 32 + th * 16);
            float4 f0 = xp[0], f1 = xp[1], f2 = xp[2], f3 = xp[3];
            v8h h0, h1;
            h0[0] = (_Float16)f0.x; h0[1] = (_Float16)f0.y;
            h0[2] = (_Float16)f0.z; h0[3] = (_Float16)f0.w;
            h0[4] = (_Float16)f1.x; h0[5] = (_Float16)f1.y;
            h0[6] = (_Float16)f1.z; h0[7] = (_Float16)f1.w;
            h1[0] = (_Float16)f2.x; h1[1] = (_Float16)f2.y;
            h1[2] = (_Float16)f2.z; h1[3] = (_Float16)f2.w;
            h1[4] = (_Float16)f3.x; h1[5] = (_Float16)f3.y;
            h1[6] = (_Float16)f3.z; h1[7] = (_Float16)f3.w;
            *(v8h*)&Xs[tr * 32 + th * 16]     = h0;
            *(v8h*)&Xs[tr * 32 + th * 16 + 8] = h1;
        }
        __syncthreads();
        // prefetch next K-chunk of the weight stream into GL2
        if (kc + 1 < 16)
            __builtin_prefetch(wpack + (((nb * 4) * 16 + kc + 1) * 32 + lane) * 16, 0, 1);
        v16h a;
        {
            int ro = (w * 16 + laneRow) * 32;
            a = join16(*(const v8h*)&Xs[ro + base0],
                       *(const v8h*)&Xs[ro + base0 + 16]);
        }
#pragma unroll
        for (int m = 0; m < 3; ++m) {
#pragma unroll
            for (int nt = 0; nt < 4; ++nt) {
                const _Float16* wp = wpack + m * CxC +
                    (((nb * 4 + nt) * 16 + kc) * 32 + lane) * 16;
                v16h bf = join16(*(const v8h*)wp, *(const v8h*)(wp + 8));
                acc[m][nt] = __builtin_amdgcn_wmma_f32_16x16x32_f16(
                    false, a, false, bf, (short)0, acc[m][nt], false, false);
            }
        }
        __syncthreads();
    }

    const int rowAdd = (lane >> 4) << 3;
#pragma unroll
    for (int m = 0; m < 3; ++m) {
#pragma unroll
        for (int nt = 0; nt < 4; ++nt) {
            int col = nb * 64 + nt * 16 + laneRow;
            float bs = biasF[m * Cc + col];
            int h = col >> 6, d = col & 63;
#pragma unroll
            for (int r = 0; r < 8; ++r) {
                int row = m0 + w * 16 + r + rowAdd;
                int tb = row >> 10, n = row & 1023;
                int hb = tb * 8 + h;
                _Float16 hv = (_Float16)(acc[m][nt][r] + bs);
                if (m == 0)      qh[hb * 65536 + n * 64 + d]  = hv;
                else if (m == 1) kT[hb * 65536 + d * 1024 + n] = hv;
                else             vh[hb * 65536 + n * 64 + d]  = hv;
            }
        }
    }
}

// ---------------------------------------------------------------------------
// Kernel 3: kv[tbh] = k^T(64x1024) @ v(1024x64) / N. One block per (t,b,h).
// 8 waves: wave = (dt 0..3 row tile) x (eh 0..1 col half), 2 accums/wave.
// Tiles staged with async global->LDS copies (ASYNCcnt).
// ---------------------------------------------------------------------------
__global__ __launch_bounds__(256)
void kv_kernel(const _Float16* __restrict__ kT, const _Float16* __restrict__ vh,
               _Float16* __restrict__ kvh) {
    __shared__ __align__(16) _Float16 Ks[64 * 32];
    __shared__ __align__(16) _Float16 Vs[32 * 64];
    const int t = threadIdx.x, w = t >> 5, lane = t & 31;
    const int tbh = blockIdx.x;
    const int dt = w & 3, eh = w >> 2;
    const int laneRow = lane & 15;
    const int base0   = (lane < 16) ? 0 : 8;
    v8f acc[2] = {};
    const _Float16* kbase = kT + tbh * 65536;
    const _Float16* vbase = vh + tbh * 65536;

    for (int nc = 0; nc < 32; ++nc) {
        int n0 = nc * 32;
        { int d = t >> 2, c0 = (t & 3) * 8;
          stage_b128(&Ks[d * 32 + c0], kbase + d * 1024 + n0 + c0); }
        { int nn = t >> 3, e0 = (t & 7) * 8;
          stage_b128(&Vs[nn * 64 + e0], vbase + (n0 + nn) * 64 + e0); }
        stage_fence();
        v16h a;
        { int ro = (dt * 16 + laneRow) * 32;
          a = join16(*(const v8h*)&Ks[ro + base0],
                     *(const v8h*)&Ks[ro + base0 + 16]); }
#pragma unroll
        for (int et = 0; et < 2; ++et) {
            int col = (eh * 2 + et) * 16 + laneRow;
            v16h b;
#pragma unroll
            for (int i = 0; i < 8; ++i) {
                b[i]     = Vs[(base0 + i) * 64 + col];
                b[8 + i] = Vs[(base0 + 16 + i) * 64 + col];
            }
            acc[et] = __builtin_amdgcn_wmma_f32_16x16x32_f16(
                false, a, false, b, (short)0, acc[et], false, false);
        }
        __syncthreads();
    }
    const int rowAdd = (lane >> 4) << 3;
#pragma unroll
    for (int et = 0; et < 2; ++et) {
#pragma unroll
        for (int r = 0; r < 8; ++r) {
            int d = dt * 16 + r + rowAdd;
            int e = (eh * 2 + et) * 16 + laneRow;
            kvh[tbh * 4096 + d * 64 + e] =
                (_Float16)(acc[et][r] * (1.0f / 1024.0f));
        }
    }
}

// ---------------------------------------------------------------------------
// Kernel 4: y[tbh] = q(1024x64) @ kv(64x64). kv tile (8KB, contiguous) is a
// textbook Tensor Data Mover job: wave 0 issues one TENSOR_LOAD_TO_LDS
// descriptor (D#: 1x4096 f16 tile), waits TENSORcnt, barrier publishes.
// ---------------------------------------------------------------------------
__global__ __launch_bounds__(256)
void y_kernel(const _Float16* __restrict__ qh, const _Float16* __restrict__ kvh,
              _Float16* __restrict__ yh) {
    __shared__ __align__(16) _Float16 KVs[64 * 64];
    const int t = threadIdx.x, w = t >> 5, lane = t & 31;
    const int mb = blockIdx.x, tbh = blockIdx.y;
    const int tb = tbh >> 3, h = tbh & 7;

#ifdef HAS_TDM
    if (t < 32) {
        unsigned ldsOff =
            (unsigned)(unsigned long long)(__attribute__((address_space(3))) _Float16*)&KVs[0];
        unsigned long long ga = (unsigned long long)(const void*)(kvh + tbh * 4096);
        const unsigned tdim0 = 4096, tile0 = 4096;     // one 4096-elem row, 2B elems
        const unsigned long long str0 = 4096;
        u32x4 g0; i32x8 g1;
        i32x4 g2 = {0, 0, 0, 0}, g3 = {0, 0, 0, 0};
        g0[0] = 1u;                                    // count=1, user descriptor
        g0[1] = ldsOff;                                // lds_addr
        g0[2] = (unsigned)(ga & 0xffffffffull);        // global_addr[31:0]
        g0[3] = (unsigned)((ga >> 32) & 0x1ffffffull) | (2u << 30);  // type=2
        g1[0] = 0x10000;                               // data_size=1 (2 bytes)
        g1[1] = (int)((tdim0 & 0xffffu) << 16);        // tensor_dim0[15:0]
        g1[2] = (int)(((tdim0 >> 16) & 0xffffu) | (1u << 16));       // dim0 hi | tensor_dim1=1
        g1[3] = (int)((tile0 & 0xffffu) << 16);        // tensor_dim1 hi=0 | tile_dim0
        g1[4] = 1;                                     // tile_dim1=1, tile_dim2=0
        g1[5] = (int)(unsigned)(str0 & 0xffffffffull); // tensor_dim0_stride lo
        g1[6] = (int)(unsigned)((str0 >> 32) & 0xffffull);
        g1[7] = 0;
#if __has_include(<hip/amd_detail/amd_gfx1250_TDM.h>)
        i32x8 g4 = {0, 0, 0, 0, 0, 0, 0, 0};
        __builtin_amdgcn_tensor_load_to_lds(g0, g1, g2, g3, g4, 0);
#else
        __builtin_amdgcn_tensor_load_to_lds(g0, g1, g2, g3, 0);
#endif
        wait_tensor0();
    }
    __syncthreads();
#else
    { int d = t >> 2, e0 = (t & 3) * 16;
      const _Float16* src = kvh + tbh * 4096 + d * 64 + e0;
      stage_b128(&KVs[d * 64 + e0], src);
      stage_b128(&KVs[d * 64 + e0 + 8], src + 8); }
    stage_fence();
#endif

    const int laneRow = lane & 15;
    const int base0   = (lane < 16) ? 0 : 8;
    v8f acc[4] = {};
#pragma unroll
    for (int kc = 0; kc < 2; ++kc) {
        const _Float16* ap = qh + tbh * 65536 +
            (mb * 128 + w * 16 + laneRow) * 64 + kc * 32 + base0;
        v16h a = join16(*(const v8h*)ap, *(const v8h*)(ap + 16));
#pragma unroll
        for (int nt = 0; nt < 4; ++nt) {
            int col = nt * 16 + laneRow;
            v16h b;
#pragma unroll
            for (int i = 0; i < 8; ++i) {
                b[i]     = KVs[(kc * 32 + base0 + i) * 64 + col];
                b[8 + i] = KVs[(kc * 32 + base0 + 16 + i) * 64 + col];
            }
            acc[nt] = __builtin_amdgcn_wmma_f32_16x16x32_f16(
                false, a, false, b, (short)0, acc[nt], false, false);
        }
    }
    const int rowAdd = (lane >> 4) << 3;
#pragma unroll
    for (int nt = 0; nt < 4; ++nt) {
#pragma unroll
        for (int r = 0; r < 8; ++r) {
            int n = mb * 128 + w * 16 + r + rowAdd;
            int e = nt * 16 + laneRow;
            yh[(tb * 1024 + n) * Cc + h * 64 + e] = (_Float16)acc[nt][r];
        }
    }
}

// ---------------------------------------------------------------------------
// Kernel 5: out = y @ Wp'^T + bp' (BN folded), fp32 output.
// y tile staged with async global->LDS copies.
// ---------------------------------------------------------------------------
__global__ __launch_bounds__(256)
void proj_kernel(const _Float16* __restrict__ yh,
                 const _Float16* __restrict__ wpack,
                 const float* __restrict__ biasF, float* __restrict__ out) {
    __shared__ __align__(16) _Float16 Ys[128 * 32];
    const int t = threadIdx.x, w = t >> 5, lane = t & 31;
    const int m0 = blockIdx.x * 128, nb = blockIdx.y;
    const int laneRow = lane & 15;
    const int base0   = (lane < 16) ? 0 : 8;
    v8f acc[4] = {};

    for (int kc = 0; kc < 16; ++kc) {
        { int tr = t >> 1, th = t & 1;
          const _Float16* sp = yh + (m0 + tr) * Cc + kc * 32 + th * 16;
          stage_b128(&Ys[tr * 32 + th * 16], sp);
          stage_b128(&Ys[tr * 32 + th * 16 + 8], sp + 8); }
        stage_fence();
        if (kc + 1 < 16)
            __builtin_prefetch(wpack + 3 * CxC +
                               (((nb * 4) * 16 + kc + 1) * 32 + lane) * 16, 0, 1);
        v16h a;
        { int ro = (w * 16 + laneRow) * 32;
          a = join16(*(const v8h*)&Ys[ro + base0],
                     *(const v8h*)&Ys[ro + base0 + 16]); }
#pragma unroll
        for (int nt = 0; nt < 4; ++nt) {
            const _Float16* wp = wpack + 3 * CxC +
                (((nb * 4 + nt) * 16 + kc) * 32 + lane) * 16;
            v16h bf = join16(*(const v8h*)wp, *(const v8h*)(wp + 8));
            acc[nt] = __builtin_amdgcn_wmma_f32_16x16x32_f16(
                false, a, false, bf, (short)0, acc[nt], false, false);
        }
        __syncthreads();
    }
    const int rowAdd = (lane >> 4) << 3;
#pragma unroll
    for (int nt = 0; nt < 4; ++nt) {
        int col = nb * 64 + nt * 16 + laneRow;
        float bs = biasF[3 * Cc + col];
#pragma unroll
        for (int r = 0; r < 8; ++r) {
            int row = m0 + w * 16 + r + rowAdd;
            out[row * Cc + col] = acc[nt][r] + bs;
        }
    }
}

// ---------------------------------------------------------------------------
extern "C" void kernel_launch(void* const* d_in, const int* in_sizes, int n_in,
                              void* d_out, int out_size, void* d_ws,
                              size_t ws_size, hipStream_t stream) {
    (void)in_sizes; (void)n_in; (void)out_size; (void)ws_size;
    const float* x = (const float*)d_in[0];
    AllParams P;
    for (int m = 0; m < 4; ++m) {
        P.W[m]    = (const float*)d_in[1 + m * 6 + 0];
        P.b[m]    = (const float*)d_in[1 + m * 6 + 1];
        P.g[m]    = (const float*)d_in[1 + m * 6 + 2];
        P.beta[m] = (const float*)d_in[1 + m * 6 + 3];
        P.mu[m]   = (const float*)d_in[1 + m * 6 + 4];
        P.var[m]  = (const float*)d_in[1 + m * 6 + 5];
    }
    char* ws = (char*)d_ws;
    _Float16* wpack = (_Float16*)(ws + OFF_WPACK);
    float*    biasF = (float*)   (ws + OFF_BIAS);
    _Float16* qh    = (_Float16*)(ws + OFF_QH);
    _Float16* kT    = (_Float16*)(ws + OFF_KT);
    _Float16* vh    = (_Float16*)(ws + OFF_VH);
    _Float16* yh    = (_Float16*)(ws + OFF_YH);
    _Float16* kvh   = (_Float16*)(ws + OFF_KVH);
    float* out = (float*)d_out;

    fold_pack_kernel<<<4096, 256, 0, stream>>>(P, wpack, biasF);
    qkv_kernel<<<dim3(256, 8), 256, 0, stream>>>(x, wpack, biasF, qh, kT, vh);
    kv_kernel<<<256, 256, 0, stream>>>(kT, vh, kvh);
    y_kernel<<<dim3(8, 256), 256, 0, stream>>>(qh, kvh, yh);
    proj_kernel<<<dim3(256, 8), 256, 0, stream>>>(yh, wpack, biasF, out);
}
